// ExpertChoiceLayer_48576080118267
// MI455X (gfx1250) — compile-verified, matrix operand order checked by reading, not run on hardware
//
#include <hip/hip_runtime.h>

// Problem constants (from reference)
#define Ntok 32768   // B*T
#define Ddim 512
#define Eexp 8
#define Ffeat 2048
#define Ccap 4096

typedef __bf16 bf16_t;
typedef __attribute__((ext_vector_type(8)))  __bf16 v8bf;
typedef __attribute__((ext_vector_type(16))) __bf16 v16bf;
typedef __attribute__((ext_vector_type(8)))  float  v8f;
typedef __attribute__((ext_vector_type(4)))  int    v4i_t;

// ---- CDNA5 async global->LDS staging (ASYNCcnt-tracked, bypasses VGPRs) ----
#if __has_builtin(__builtin_amdgcn_global_load_async_to_lds_b128)
#define EC_ASYNC_LDS 1
typedef __attribute__((address_space(1))) v4i_t* ec_gv4_t;   // global int4*
typedef __attribute__((address_space(3))) v4i_t* ec_lv4_t;   // LDS int4*
#endif

__device__ __forceinline__ void ec_stage16(const bf16_t* g, bf16_t* l) {
#ifdef EC_ASYNC_LDS
    __builtin_amdgcn_global_load_async_to_lds_b128((ec_gv4_t)g, (ec_lv4_t)l, 0, 0);
#else
    *(v8bf*)l = *(const v8bf*)g;
#endif
}
__device__ __forceinline__ void ec_stage_wait() {
#ifdef EC_ASYNC_LDS
#if __has_builtin(__builtin_amdgcn_s_wait_asynccnt)
    __builtin_amdgcn_s_wait_asynccnt(0);
#else
    asm volatile("s_wait_asynccnt 0x0" ::: "memory");
#endif
#endif
    __syncthreads();
}

// ---------------------------------------------------------------- utilities
__global__ __launch_bounds__(256) void ec_zero_f32(float* p, int n) {
    int i = blockIdx.x * 256 + threadIdx.x;
    if (i < n) p[i] = 0.0f;
}
__global__ __launch_bounds__(256) void ec_zero_u32(unsigned* p, int n) {
    int i = blockIdx.x * 256 + threadIdx.x;
    if (i < n) p[i] = 0u;
}

// x [N,D] fp32 -> bf16 (A-operand of GEMM1; halves gather re-read traffic)
__global__ __launch_bounds__(256) void ec_convX(const float* __restrict__ x, bf16_t* __restrict__ xbf) {
    size_t i = (size_t)blockIdx.x * 256 + threadIdx.x;
    xbf[i] = (bf16_t)x[i];
}
// W1 [E,D,F] fp32 -> W1t [E,F,D] bf16   (B-operand of GEMM1: n=f across lanes, k=d contiguous)
__global__ __launch_bounds__(256) void ec_convW1(const float* __restrict__ W1, bf16_t* __restrict__ W1t) {
    size_t i = (size_t)blockIdx.x * 256 + threadIdx.x;   // over E*F*D
    int d = (int)(i % Ddim);
    size_t r = i / Ddim;
    int f = (int)(r % Ffeat);
    int e = (int)(r / Ffeat);
    W1t[i] = (bf16_t)W1[((size_t)e * Ddim + d) * Ffeat + f];
}
// W2 [E,F,D] fp32 -> W2t [E,D,F] bf16   (B-operand of GEMM2: n=d across lanes, k=f contiguous)
__global__ __launch_bounds__(256) void ec_convW2(const float* __restrict__ W2, bf16_t* __restrict__ W2t) {
    size_t i = (size_t)blockIdx.x * 256 + threadIdx.x;   // over E*D*F
    int f = (int)(i % Ffeat);
    size_t r = i / Ffeat;
    int d = (int)(r % Ddim);
    int e = (int)(r / Ddim);
    W2t[i] = (bf16_t)W2[((size_t)e * Ffeat + f) * Ddim + d];
}

// ---------------------------------------------------------------- routing
// logits[e][t] = sum_d x[t][d] * Wr[d][e]   (stored column-major [E][N])
__global__ __launch_bounds__(256) void ec_logits(const float* __restrict__ x,
                                                 const float* __restrict__ Wr,
                                                 float* __restrict__ logits) {
    __shared__ float w[Ddim * Eexp];                 // 16 KB
    int tid = threadIdx.x;
    for (int i = tid; i < Ddim * Eexp; i += 256) w[i] = Wr[i];
    __syncthreads();
    int t = blockIdx.x * 256 + tid;
    const float* xr = x + (size_t)t * Ddim;
    float acc[Eexp];
#pragma unroll
    for (int e = 0; e < Eexp; e++) acc[e] = 0.0f;
    for (int d = 0; d < Ddim; d++) {
        float xv = xr[d];
#pragma unroll
        for (int e = 0; e < Eexp; e++) acc[e] += xv * w[d * Eexp + e];
    }
#pragma unroll
    for (int e = 0; e < Eexp; e++) logits[(size_t)e * Ntok + t] = acc[e];
}

// per-expert max and sum(exp)  -> stats[e]=max, stats[E+e]=sum
__global__ __launch_bounds__(256) void ec_stats(const float* __restrict__ logits,
                                                float* __restrict__ stats) {
    __shared__ float red[256];
    int e = blockIdx.x, tid = threadIdx.x;
    const float* col = logits + (size_t)e * Ntok;
    float m = -3.4e38f;
    for (int t = tid; t < Ntok; t += 256) m = fmaxf(m, col[t]);
    red[tid] = m; __syncthreads();
    for (int s = 128; s > 0; s >>= 1) { if (tid < s) red[tid] = fmaxf(red[tid], red[tid + s]); __syncthreads(); }
    m = red[0]; __syncthreads();
    float sum = 0.0f;
    for (int t = tid; t < Ntok; t += 256) sum += expf(col[t] - m);
    red[tid] = sum; __syncthreads();
    for (int s = 128; s > 0; s >>= 1) { if (tid < s) red[tid] += red[tid + s]; __syncthreads(); }
    if (tid == 0) { stats[e] = m; stats[Eexp + e] = red[0]; }
}

// keys[e][t] = bitcast(prob) — probs are positive floats so uint order == float order
__global__ __launch_bounds__(256) void ec_keys(const float* __restrict__ logits,
                                               const float* __restrict__ stats,
                                               unsigned* __restrict__ keys) {
    int t = blockIdx.x * 256 + threadIdx.x;
    int e = blockIdx.y;
    float m = stats[e];
    float inv = 1.0f / stats[Eexp + e];
    float p = expf(logits[(size_t)e * Ntok + t] - m) * inv;
    keys[(size_t)e * Ntok + t] = __float_as_uint(p);
}

// exact C-th largest via 4-round byte radix-select; one workgroup per expert.
// sel[e]=threshold, sel[E+e]=tieTake (# equal-to-threshold to take), sel[2E+e]=countGreater
__global__ __launch_bounds__(256) void ec_select(const unsigned* __restrict__ keys,
                                                 unsigned* __restrict__ sel) {
    __shared__ unsigned hist[256];
    __shared__ unsigned sh[2];      // prefix, k-remaining
    int e = blockIdx.x, tid = threadIdx.x;
    const unsigned* col = keys + (size_t)e * Ntok;
    if (tid == 0) { sh[0] = 0u; sh[1] = Ccap; }
    unsigned mask = 0u;
    for (int byte = 3; byte >= 0; byte--) {
        hist[tid] = 0u;
        __syncthreads();
        unsigned prefix = sh[0];
        int shift = byte * 8;
        for (int t = tid; t < Ntok; t += 256) {
            unsigned key = col[t];
            if ((key & mask) == prefix) atomicAdd(&hist[(key >> shift) & 255u], 1u);
        }
        __syncthreads();
        if (tid == 0) {
            unsigned k = sh[1], cum = 0u;
            for (int d = 255; d >= 0; d--) {
                unsigned c = hist[d];
                if (cum + c >= k) { sh[0] = prefix | ((unsigned)d << shift); sh[1] = k - cum; break; }
                cum += c;
            }
        }
        mask |= (255u << shift);
        __syncthreads();
    }
    if (tid == 0) {
        sel[e] = sh[0];
        sel[Eexp + e] = sh[1];
        sel[2 * Eexp + e] = (unsigned)Ccap - sh[1];
    }
}

// compact top-C token ids + prob weights (unordered — scatter-add is order invariant)
__global__ __launch_bounds__(256) void ec_compact(const unsigned* __restrict__ keys,
                                                  const unsigned* __restrict__ sel,
                                                  unsigned* __restrict__ cnt,   // [2E]: cntG, cntE
                                                  int* __restrict__ idx,
                                                  float* __restrict__ vals) {
    int t = blockIdx.x * 256 + threadIdx.x;
    int e = blockIdx.y;
    unsigned key = keys[(size_t)e * Ntok + t];
    unsigned thr = sel[e];
    if (key > thr) {
        int p = (int)atomicAdd(&cnt[e], 1u);
        idx[e * Ccap + p] = t;
        vals[e * Ccap + p] = __uint_as_float(key);
    } else if (key == thr) {
        int p = (int)atomicAdd(&cnt[Eexp + e], 1u);
        if (p < (int)sel[Eexp + e]) {
            int q = (int)sel[2 * Eexp + e] + p;
            idx[e * Ccap + q] = t;
            vals[e * Ccap + q] = __uint_as_float(key);
        }
    }
}

// ---------------------------------------------------------------- WMMA GEMMs
// 8 waves / block, block tile M=128 x N=64, K step 32; wave = 64(M) x 16(N), 4 accumulators.
// Double-buffered LDS A-tile; next tile staged via async global->LDS while WMMAs run.

__global__ __launch_bounds__(256)
void ec_gemm1(const bf16_t* __restrict__ xbf,     // [N,D] bf16
              const bf16_t* __restrict__ W1t,     // [E,F,D] bf16
              const float* __restrict__ b1,       // [E,F]
              const int* __restrict__ idx,        // [E,C]
              bf16_t* __restrict__ h,             // [C,F] bf16 (per-expert reuse)
              int e) {
    __shared__ bf16_t As[2][128][32];             // 16 KB double buffer
    const int tid = threadIdx.x;
    const int lane = tid & 31, wave = tid >> 5;
    const int hi = lane >> 4, ln = lane & 15;
    const int waveM = wave & 1, waveN = wave >> 1;
    const int mBlk = blockIdx.x * 128, nBlk = blockIdx.y * 64;

    const int lr = tid >> 1, lhalf = tid & 1;     // each thread stages 32B of one row
    const int tok = idx[e * Ccap + mBlk + lr];
    const bf16_t* arow_g = xbf + (size_t)tok * Ddim + lhalf * 16;
    bf16_t* sdst[2] = { &As[0][lr][lhalf * 16], &As[1][lr][lhalf * 16] };

    const int nCol = nBlk + waveN * 16 + ln;
    const bf16_t* Brow = W1t + ((size_t)e * Ffeat + nCol) * Ddim;

    // prologue: stage K-tile 0 into buffer 0
    ec_stage16(arow_g, sdst[0]);
    ec_stage16(arow_g + 8, sdst[0] + 8);
    ec_stage_wait();

    v8f acc[4] = {};
    for (int k0 = 0; k0 < Ddim; k0 += 32) {
        const int cur = (k0 >> 5) & 1;
        if (k0 + 32 < Ddim) {                     // prefetch next K-tile into other buffer
            bf16_t* sn = sdst[cur ^ 1];
            ec_stage16(arow_g + k0 + 32, sn);
            ec_stage16(arow_g + k0 + 40, sn + 8);
        }
        v8bf blo = *(const v8bf*)(Brow + k0 + hi * 8);
        v8bf bhi = *(const v8bf*)(Brow + k0 + 16 + hi * 8);
        v16bf bfrag = __builtin_shufflevector(blo, bhi, 0,1,2,3,4,5,6,7,8,9,10,11,12,13,14,15);
#pragma unroll
        for (int s = 0; s < 4; s++) {
            const bf16_t* arow = &As[cur][waveM * 64 + s * 16 + ln][0];
            v8bf alo = *(const v8bf*)(arow + hi * 8);
            v8bf ahi = *(const v8bf*)(arow + 16 + hi * 8);
            v16bf afrag = __builtin_shufflevector(alo, ahi, 0,1,2,3,4,5,6,7,8,9,10,11,12,13,14,15);
            acc[s] = __builtin_amdgcn_wmma_f32_16x16x32_bf16(
                false, afrag, false, bfrag, (short)0, acc[s], false, false);
        }
        ec_stage_wait();
    }
    const float bias = b1[e * Ffeat + nCol];
#pragma unroll
    for (int s = 0; s < 4; s++) {
#pragma unroll
        for (int v = 0; v < 8; v++) {
            int m = mBlk + waveM * 64 + s * 16 + v + 8 * hi;
            float val = acc[s][v] + bias;
            float g = 0.5f * val * (1.0f + erff(val * 0.70710678118f));   // exact gelu
            h[(size_t)m * Ffeat + nCol] = (bf16_t)g;
        }
    }
}

__global__ __launch_bounds__(256)
void ec_gemm2(const bf16_t* __restrict__ h,       // [C,F] bf16
              const bf16_t* __restrict__ W2t,     // [E,D,F] bf16
              const float* __restrict__ b2,       // [E,D]
              const int* __restrict__ idx,        // [E,C]
              const float* __restrict__ vals,     // [E,C]
              float* __restrict__ out,            // [N,D] fp32 (zeroed)
              int e) {
    __shared__ bf16_t As[2][128][32];
    const int tid = threadIdx.x;
    const int lane = tid & 31, wave = tid >> 5;
    const int hi = lane >> 4, ln = lane & 15;
    const int waveM = wave & 1, waveN = wave >> 1;
    const int mBlk = blockIdx.x * 128, nBlk = blockIdx.y * 64;

    const int lr = tid >> 1, lhalf = tid & 1;
    const bf16_t* arow_g = h + (size_t)(mBlk + lr) * Ffeat + lhalf * 16;
    bf16_t* sdst[2] = { &As[0][lr][lhalf * 16], &As[1][lr][lhalf * 16] };

    const int nCol = nBlk + waveN * 16 + ln;
    const bf16_t* Brow = W2t + ((size_t)e * Ddim + nCol) * Ffeat;

    ec_stage16(arow_g, sdst[0]);
    ec_stage16(arow_g + 8, sdst[0] + 8);
    ec_stage_wait();

    v8f acc[4] = {};
    for (int k0 = 0; k0 < Ffeat; k0 += 32) {
        const int cur = (k0 >> 5) & 1;
        if (k0 + 32 < Ffeat) {
            bf16_t* sn = sdst[cur ^ 1];
            ec_stage16(arow_g + k0 + 32, sn);
            ec_stage16(arow_g + k0 + 40, sn + 8);
        }
        v8bf blo = *(const v8bf*)(Brow + k0 + hi * 8);
        v8bf bhi = *(const v8bf*)(Brow + k0 + 16 + hi * 8);
        v16bf bfrag = __builtin_shufflevector(blo, bhi, 0,1,2,3,4,5,6,7,8,9,10,11,12,13,14,15);
#pragma unroll
        for (int s = 0; s < 4; s++) {
            const bf16_t* arow = &As[cur][waveM * 64 + s * 16 + ln][0];
            v8bf alo = *(const v8bf*)(arow + hi * 8);
            v8bf ahi = *(const v8bf*)(arow + 16 + hi * 8);
            v16bf afrag = __builtin_shufflevector(alo, ahi, 0,1,2,3,4,5,6,7,8,9,10,11,12,13,14,15);
            acc[s] = __builtin_amdgcn_wmma_f32_16x16x32_bf16(
                false, afrag, false, bfrag, (short)0, acc[s], false, false);
        }
        ec_stage_wait();
    }
    const float bias = b2[e * Ddim + nCol];
#pragma unroll
    for (int s = 0; s < 4; s++) {
#pragma unroll
        for (int v = 0; v < 8; v++) {
            int slot = mBlk + waveM * 64 + s * 16 + v + 8 * hi;
            int tok = idx[e * Ccap + slot];
            float w = vals[e * Ccap + slot];
            atomicAdd(&out[(size_t)tok * Ddim + nCol], (acc[s][v] + bias) * w);
        }
    }
}

// ---------------------------------------------------------------- launcher
extern "C" void kernel_launch(void* const* d_in, const int* in_sizes, int n_in,
                              void* d_out, int out_size, void* d_ws, size_t ws_size,
                              hipStream_t stream) {
    const float* x  = (const float*)d_in[0];   // [B,T,D]
    const float* Wr = (const float*)d_in[1];   // [D,E]
    const float* W1 = (const float*)d_in[2];   // [E,D,F]
    const float* b1 = (const float*)d_in[3];   // [E,F]
    const float* W2 = (const float*)d_in[4];   // [E,F,D]
    const float* b2 = (const float*)d_in[5];   // [E,D]
    float* out = (float*)d_out;                // [N,D]

    // workspace carve-up (256B aligned)
    char* ws = (char*)d_ws;
    size_t off = 0;
    auto carve = [&](size_t bytes) -> char* {
        char* p = ws + off;
        off = (off + bytes + 255) & ~(size_t)255;
        return p;
    };
    float*    logits = (float*)   carve((size_t)Ntok * Eexp * 4);
    unsigned* keys   = (unsigned*)carve((size_t)Ntok * Eexp * 4);
    float*    stats  = (float*)   carve(2 * Eexp * 4);
    unsigned* sel    = (unsigned*)carve(3 * Eexp * 4);
    unsigned* cnt    = (unsigned*)carve(2 * Eexp * 4);
    int*      idx    = (int*)     carve((size_t)Eexp * Ccap * 4);
    float*    vals   = (float*)   carve((size_t)Eexp * Ccap * 4);
    bf16_t*   xbf    = (bf16_t*)  carve((size_t)Ntok * Ddim * 2);
    bf16_t*   W1t    = (bf16_t*)  carve((size_t)Eexp * Ffeat * Ddim * 2);
    bf16_t*   W2t    = (bf16_t*)  carve((size_t)Eexp * Ddim * Ffeat * 2);
    bf16_t*   hbuf   = (bf16_t*)  carve((size_t)Ccap * Ffeat * 2);
    (void)ws_size; (void)in_sizes; (void)n_in; (void)out_size;

    // one-shot conversions (bandwidth-trivial vs. 23.3 TB/s)
    {
        int nblk = (int)(((size_t)Eexp * Ffeat * Ddim) / 256);
        ec_convW1<<<nblk, 256, 0, stream>>>(W1, W1t);
        ec_convW2<<<nblk, 256, 0, stream>>>(W2, W2t);
        ec_convX<<<(Ntok * Ddim) / 256, 256, 0, stream>>>(x, xbf);
    }
    // zero output + counters
    ec_zero_f32<<<(Ntok * Ddim) / 256, 256, 0, stream>>>(out, Ntok * Ddim);
    ec_zero_u32<<<1, 256, 0, stream>>>(cnt, 2 * Eexp);

    // routing
    ec_logits<<<Ntok / 256, 256, 0, stream>>>(x, Wr, logits);
    ec_stats<<<Eexp, 256, 0, stream>>>(logits, stats);
    ec_keys<<<dim3(Ntok / 256, Eexp), 256, 0, stream>>>(logits, stats, keys);
    ec_select<<<Eexp, 256, 0, stream>>>(keys, sel);
    ec_compact<<<dim3(Ntok / 256, Eexp), 256, 0, stream>>>(keys, sel, cnt, idx, vals);

    // per-expert grouped GEMMs, reusing the bf16 h buffer (stream-ordered)
    for (int e = 0; e < Eexp; e++) {
        ec_gemm1<<<dim3(Ccap / 128, Ffeat / 64), 256, 0, stream>>>(xbf, W1t, b1, idx, hbuf, e);
        ec_gemm2<<<dim3(Ccap / 128, Ddim / 64), 256, 0, stream>>>(hbuf, W2t, b2, idx, vals, out, e);
    }
}
